// Stage_55250459296226
// MI455X (gfx1250) — matrix operand order repaired and checked by reading, not compiled, not run on hardware
//
#include <hip/hip_runtime.h>
#include <cstdint>

// ---------------------------------------------------------------------------
// Types (trivial ext-vector types so they can live in unions)
// ---------------------------------------------------------------------------
typedef __attribute__((ext_vector_type(16))) _Float16 v16h;
typedef __attribute__((ext_vector_type(8)))  float    v8f;
typedef __attribute__((ext_vector_type(4)))  unsigned int u32x4;
typedef __attribute__((ext_vector_type(4)))  int      i32x4;
typedef __attribute__((ext_vector_type(2)))  float    f32x2;

union AU { v16h h; _Float16 e[16]; u32x4 q[2]; };

#define BN_INV 0.9999950000374997f   // 1/sqrt(1+1e-5)
#define NPTS   32768
#define KNN    16
#define DIM    128

#define AS1 __attribute__((address_space(1)))
#define AS3 __attribute__((address_space(3)))

__device__ __forceinline__ float gelu_f(float x){
  return 0.5f * x * (1.0f + erff(x * 0.70710678118654752440f));
}

__device__ __forceinline__ v8f wmma_f16(v16h a, v16h b, v8f c){
  // D = A(16x32 f16) * B(32x16 f16) + C(16x16 f32)
  return __builtin_amdgcn_wmma_f32_16x16x32_f16(false, a, false, b, (short)0, c, false, false);
}

__device__ __forceinline__ v8f splat8(float x){
  v8f r;
  #pragma unroll
  for (int i = 0; i < 8; ++i) r[i] = x;
  return r;
}

// ---------------------------------------------------------------------------
// Async global->LDS staging (gfx1250 GLOBAL_LOAD_ASYNC_TO_LDS_B128, ASYNCcnt),
// with a synchronous fallback if the builtin is unavailable on this toolchain.
// Builtin signature (from hipcc diagnostic): (int4 AS1*, int4 AS3*, Imm, Imm).
// ---------------------------------------------------------------------------
__device__ __forceinline__ void stage_async(const uint32_t* __restrict__ g,
                                            uint32_t* l, int dwords, int tid, int nthr){
#if __has_builtin(__builtin_amdgcn_global_load_async_to_lds_b128)
  for (int i = tid * 4; i < dwords; i += nthr * 4){
    __builtin_amdgcn_global_load_async_to_lds_b128(
        (AS1 i32x4*)(g + i), (AS3 i32x4*)(l + i), 0, 0);
  }
#else
  for (int i = tid * 4; i < dwords; i += nthr * 4)
    *(u32x4*)(l + i) = *(const u32x4*)(g + i);
#endif
}

__device__ __forceinline__ void stage_wait(){
#if __has_builtin(__builtin_amdgcn_global_load_async_to_lds_b128)
# if __has_builtin(__builtin_amdgcn_s_wait_asynccnt)
  __builtin_amdgcn_s_wait_asynccnt(0);
# else
  asm volatile("s_wait_asynccnt 0x0" ::: "memory");
# endif
#endif
  __syncthreads();
}

// B operand: repacked f16 blocks. Block (kc,j): 32 lanes x 8 dwords, lane-contiguous.
// Works for both global and LDS base pointers.
__device__ __forceinline__ v16h loadB(const uint32_t* __restrict__ w, int kc, int j, int NT, int lane){
  const u32x4* p = (const u32x4*)(w + (size_t)((((kc * NT + j) << 5) + lane) << 3));
  AU u; u.q[0] = p[0]; u.q[1] = p[1];
  return u.h;
}

// Software-pipelined 8-tile WMMA pass: acc[j] = A * B[kc, jBase+j] + acc[j].
// Double-buffers the B operand so each ds/vmem load pair is covered by a WMMA.
__device__ __forceinline__ void gemm8(v16h a, const uint32_t* __restrict__ w,
                                      int kc, int jBase, int NT, int lane, v8f* acc){
  v16h b = loadB(w, kc, jBase, NT, lane);
  #pragma unroll
  for (int j = 0; j < 8; ++j){
    v16h bnext = (j < 7) ? loadB(w, kc, jBase + j + 1, NT, lane) : b;
    acc[j] = wmma_f16(a, b, acc[j]);
    b = bnext;
  }
}

// A operand from a row of f32 data (row = row0 + lane%16), converted on the fly.
// A layout: lane group g, reg v, half -> K = kc*32 + 16*(v>>2) + 8g + 2*(v&3) + half
__device__ __forceinline__ v16h loadA_f32(const float* __restrict__ rowp, int kc, int g){
  AU u;
  #pragma unroll
  for (int v = 0; v < 8; ++v){
    int k = kc * 32 + ((v >> 2) << 4) + (g << 3) + ((v & 3) << 1);
    f32x2 f = *(const f32x2*)(rowp + k);
    u.e[2 * v]     = (_Float16)f.x;
    u.e[2 * v + 1] = (_Float16)f.y;
  }
  return u.h;
}

// A operand with fused BatchNorm applied to the f32 row.
__device__ __forceinline__ v16h loadA_bn(const float* __restrict__ rowp,
                                         const float* __restrict__ sS,
                                         const float* __restrict__ sT, int kc, int g){
  AU u;
  #pragma unroll
  for (int v = 0; v < 8; ++v){
    int k = kc * 32 + ((v >> 2) << 4) + (g << 3) + ((v & 3) << 1);
    f32x2 f = *(const f32x2*)(rowp + k);
    u.e[2 * v]     = (_Float16)(f.x * sS[k]     + sT[k]);
    u.e[2 * v + 1] = (_Float16)(f.y * sS[k + 1] + sT[k + 1]);
  }
  return u.h;
}

// A operand from the per-wave LDS staging tile (16 rows x 128 f16, stride 136).
__device__ __forceinline__ v16h loadA_stage(const _Float16* __restrict__ s, int kc, int lane){
  int g = lane >> 4, r = lane & 15;
  const u32x4* p0 = (const u32x4*)(s + r * 136 + kc * 32 + (g << 3));
  const u32x4* p1 = (const u32x4*)(s + r * 136 + kc * 32 + 16 + (g << 3));
  AU u; u.q[0] = *p0; u.q[1] = *p1;
  return u.h;
}

// ---------------------------------------------------------------------------
// Weight repack: f32 [Kd x Nd] row-major -> f16 WMMA B-operand blocks.
// ---------------------------------------------------------------------------
__global__ void repack_b16(const float* __restrict__ W, uint32_t* __restrict__ out, int Kd, int Nd){
  int t = blockIdx.x * blockDim.x + threadIdx.x;
  int total = (Kd * Nd) >> 1;
  if (t >= total) return;
  int v    = t & 7;
  int lane = (t >> 3) & 31;
  int NT   = Nd >> 4;
  int j    = (t >> 8) % NT;
  int kc   = t / (NT << 8);
  int k = kc * 32 + ((v >> 2) << 4) + ((lane >> 4) << 3) + ((v & 3) << 1);   // B layout: K = 16*(v>>2) + 8*(lane>>4) + 2*(v&3) + half
  int n = j * 16 + (lane & 15);
  union { uint32_t u; _Float16 e[2]; } pk;
  pk.e[0] = (_Float16)W[(size_t)k * Nd + n];
  pk.e[1] = (_Float16)W[(size_t)(k + 1) * Nd + n];
  out[t] = pk.u;
}

// ---------------------------------------------------------------------------
// Fuse p1 = (dxyz@w1 + b1)@w2 + b2  ->  dxyz@w12 + b12   (per depth)
// ---------------------------------------------------------------------------
__global__ void fuse_w12(const float* __restrict__ w1, const float* __restrict__ b1,
                         const float* __restrict__ w2, const float* __restrict__ b2,
                         float* __restrict__ w12, float* __restrict__ b12){
  int t = threadIdx.x;         // 256 threads, single block
  int d = t >> 6, j = t & 63;
  const float* W1 = w1 + d * 192;
  const float* W2 = w2 + d * 4096;
  #pragma unroll
  for (int ci = 0; ci < 3; ++ci){
    float s = 0.f;
    for (int m = 0; m < 64; ++m) s += W1[ci * 64 + m] * W2[m * 64 + j];
    w12[d * 192 + ci * 64 + j] = s;
  }
  float s = b2[d * 64 + j];
  for (int m = 0; m < 64; ++m) s += b1[d * 64 + m] * W2[m * 64 + j];
  b12[d * 64 + j] = s;
}

// ---------------------------------------------------------------------------
// Neighbor embedding: 10 -> 16 -> 32 -> 128 MLP, max-pool over K, BN.
// Also writes dxyz[N,K,3]. One thread per point, output in 4 chunks of 32.
// ---------------------------------------------------------------------------
__global__ __launch_bounds__(256) void nbr_kernel(
    const float* __restrict__ x, const float* __restrict__ xyz, const int* __restrict__ knn,
    const float* __restrict__ w1, const float* __restrict__ g1, const float* __restrict__ b1,
    const float* __restrict__ w2, const float* __restrict__ g2, const float* __restrict__ b2,
    const float* __restrict__ w3, const float* __restrict__ ng, const float* __restrict__ nb,
    float* __restrict__ feat, float* __restrict__ dxyz)
{
  __shared__ float sw[5120];   // w1 160 | g1 16 | b1 16 | w2 512 | g2 32 | b2 32 | w3 4096 | ng 128 | nb 128
  const int W1o=0, G1o=160, B1o=176, W2o=192, G2o=704, B2o=736, W3o=768, NGo=4864, NBo=4992;
  for (int t = threadIdx.x; t < 160;  t += 256) sw[W1o+t] = w1[t];
  for (int t = threadIdx.x; t < 16;   t += 256){ sw[G1o+t] = g1[t]; sw[B1o+t] = b1[t]; }
  for (int t = threadIdx.x; t < 512;  t += 256) sw[W2o+t] = w2[t];
  for (int t = threadIdx.x; t < 32;   t += 256){ sw[G2o+t] = g2[t]; sw[B2o+t] = b2[t]; }
  for (int t = threadIdx.x; t < 4096; t += 256) sw[W3o+t] = w3[t];
  for (int t = threadIdx.x; t < 128;  t += 256){ sw[NGo+t] = ng[t]; sw[NBo+t] = nb[t]; }
  __syncthreads();

  int n = blockIdx.x * 256 + threadIdx.x;
  float px = xyz[n*3+0], py = xyz[n*3+1], pz = xyz[n*3+2];
  int nbrs[16];
  #pragma unroll
  for (int k = 0; k < 16; ++k) nbrs[k] = knn[n * KNN + k];

  for (int ch = 0; ch < 4; ++ch){
    float accm[32];
    #pragma unroll
    for (int cc = 0; cc < 32; ++cc) accm[cc] = -3.4e38f;
    for (int k = 0; k < 16; ++k){
      int nbp = nbrs[k];
      float xin[10];
      xin[0] = xyz[nbp*3+0] - px; xin[1] = xyz[nbp*3+1] - py; xin[2] = xyz[nbp*3+2] - pz;
      if (ch == 0){
        dxyz[(n * KNN + k)*3 + 0] = xin[0];
        dxyz[(n * KNN + k)*3 + 1] = xin[1];
        dxyz[(n * KNN + k)*3 + 2] = xin[2];
      }
      #pragma unroll
      for (int i = 0; i < 7; ++i) xin[3 + i] = x[nbp * 7 + i];
      float h16[16];
      #pragma unroll
      for (int o = 0; o < 16; ++o){
        float s = 0.f;
        #pragma unroll
        for (int i = 0; i < 10; ++i) s += xin[i] * sw[W1o + i*16 + o];
        h16[o] = gelu_f(s * sw[G1o + o] * BN_INV + sw[B1o + o]);
      }
      float h32[32];
      #pragma unroll
      for (int o = 0; o < 32; ++o){
        float s = 0.f;
        #pragma unroll
        for (int m = 0; m < 16; ++m) s += h16[m] * sw[W2o + m*32 + o];
        h32[o] = gelu_f(s * sw[G2o + o] * BN_INV + sw[B2o + o]);
      }
      #pragma unroll
      for (int cc = 0; cc < 32; ++cc){
        int o = ch * 32 + cc;
        float s = 0.f;
        #pragma unroll
        for (int m = 0; m < 32; ++m) s += h32[m] * sw[W3o + m*128 + o];
        accm[cc] = fmaxf(accm[cc], s);
      }
    }
    #pragma unroll
    for (int cc = 0; cc < 32; ++cc){
      int o = ch * 32 + cc;
      feat[n * DIM + o] = accm[cc] * sw[NGo + o] * BN_INV + sw[NBo + o];
    }
  }
}

// ---------------------------------------------------------------------------
// xp = feat @ proj   ([N,128]x[128,128]), one wave per 16 rows.
// Weights async-staged into dynamic LDS, shared by the block's 8 waves.
// ---------------------------------------------------------------------------
__global__ __launch_bounds__(256) void xp_kernel(const float* __restrict__ feat,
                                                 const uint32_t* __restrict__ wp,
                                                 float* __restrict__ xp)
{
  extern __shared__ __align__(16) uint32_t swp[];   // 8192 dwords = 32 KB
  stage_async(wp, swp, 8192, threadIdx.x, 256);

  int wave = threadIdx.x >> 5, lane = threadIdx.x & 31, g = lane >> 4, c = lane & 15;
  int row0 = (blockIdx.x * 8 + wave) * 16;
  const float* rowp = feat + (size_t)(row0 + c) * DIM;
  stage_wait();

  v8f acc[8];
  #pragma unroll
  for (int j = 0; j < 8; ++j) acc[j] = splat8(0.f);
  #pragma unroll
  for (int kc = 0; kc < 4; ++kc){
    v16h a = loadA_f32(rowp, kc, g);
    gemm8(a, swp, kc, 0, 8, lane, acc);
  }
  #pragma unroll
  for (int j = 0; j < 8; ++j)
    #pragma unroll
    for (int v = 0; v < 8; ++v)
      xp[(size_t)(row0 + v + 8*g) * DIM + 16*j + c] = acc[j][v];
}

// ---------------------------------------------------------------------------
// feat += g_pos @ gpe_w   ([N,64]x[64,128])
// ---------------------------------------------------------------------------
__global__ __launch_bounds__(256) void gpe_kernel(const float* __restrict__ gpos,
                                                  const uint32_t* __restrict__ wp,
                                                  float* __restrict__ feat)
{
  int wave = threadIdx.x >> 5, lane = threadIdx.x & 31, g = lane >> 4, c = lane & 15;
  int row0 = (blockIdx.x * 8 + wave) * 16;
  const float* rowp = gpos + (size_t)(row0 + c) * 64;
  v8f acc[8];
  #pragma unroll
  for (int j = 0; j < 8; ++j) acc[j] = splat8(0.f);
  #pragma unroll
  for (int kc = 0; kc < 2; ++kc){
    v16h a = loadA_f32(rowp, kc, g);
    gemm8(a, wp, kc, 0, 8, lane, acc);
  }
  #pragma unroll
  for (int j = 0; j < 8; ++j)
    #pragma unroll
    for (int v = 0; v < 8; ++v)
      feat[(size_t)(row0 + v + 8*g) * DIM + 16*j + c] += acc[j][v];
}

// ---------------------------------------------------------------------------
// feat += BN(GELU(feat@w1 + b1)@w2)   (128 -> 256 -> 128), fused two-GEMM.
// ---------------------------------------------------------------------------
__global__ __launch_bounds__(256) void mlp_kernel(
    float* __restrict__ feat, const uint32_t* __restrict__ w1p, const float* __restrict__ b1,
    const uint32_t* __restrict__ w2p, const float* __restrict__ gamma, const float* __restrict__ beta)
{
  __shared__ __align__(16) _Float16 stg[8][2176];   // 16 rows x 136 f16 per wave
  int wave = threadIdx.x >> 5, lane = threadIdx.x & 31, g = lane >> 4, c = lane & 15;
  int row0 = (blockIdx.x * 8 + wave) * 16;
  const float* rowp = feat + (size_t)(row0 + c) * DIM;

  v8f acc2[8];
  #pragma unroll
  for (int j = 0; j < 8; ++j) acc2[j] = splat8(0.f);

  for (int half = 0; half < 2; ++half){
    v8f acc1[8];
    #pragma unroll
    for (int j = 0; j < 8; ++j) acc1[j] = splat8(b1[half*128 + 16*j + c]);
    #pragma unroll
    for (int kc = 0; kc < 4; ++kc){
      v16h a = loadA_f32(rowp, kc, g);
      gemm8(a, w1p, kc, half*8, 16, lane, acc1);
    }
    #pragma unroll
    for (int j = 0; j < 8; ++j)
      #pragma unroll
      for (int v = 0; v < 8; ++v)
        stg[wave][(v + 8*g)*136 + 16*j + c] = (_Float16)gelu_f(acc1[j][v]);
    __syncthreads();
    #pragma unroll
    for (int kc = 0; kc < 4; ++kc){
      v16h a = loadA_stage(&stg[wave][0], kc, lane);
      gemm8(a, w2p, half*4 + kc, 0, 8, lane, acc2);
    }
    __syncthreads();
  }
  #pragma unroll
  for (int j = 0; j < 8; ++j){
    int col = 16*j + c;
    float s = gamma[col] * BN_INV, t = beta[col];
    #pragma unroll
    for (int v = 0; v < 8; ++v)
      feat[(size_t)(row0 + v + 8*g) * DIM + col] += acc2[j][v] * s + t;
  }
}

// ---------------------------------------------------------------------------
// LFA layer: one wave per point (16-row tile = the K=16 neighbors).
// pe = GELU(pf@w3a + b3a)@w3b + b3b ; feat += BN(max_k(pe + xp[knn]-xp))
// w3a/w3b async-staged to dynamic LDS and reused by all 8 waves.
// ---------------------------------------------------------------------------
__global__ __launch_bounds__(256) void lfa_kernel(
    float* __restrict__ feat, const float* __restrict__ xp,
    const float* __restrict__ dxyz, const int* __restrict__ knn,
    const float* __restrict__ w1, const float* __restrict__ b1,     // raw 3x64, 64 (layer slice)
    const float* __restrict__ w12, const float* __restrict__ b12,   // fused 3x64, 64
    const uint32_t* __restrict__ w3a, const float* __restrict__ b3a,
    const uint32_t* __restrict__ w3b, const float* __restrict__ b3b,
    const float* __restrict__ gamma, const float* __restrict__ beta)
{
  extern __shared__ __align__(16) uint32_t dynw[];  // 16384 dwords: w3a | w3b (64 KB)
  uint32_t* sw3a = dynw;
  uint32_t* sw3b = dynw + 8192;

  __shared__ __align__(16) _Float16 stg[8][2176];
  __shared__ float pl[8][64];     // p_local per wave
  __shared__ float dz[8][48];     // dxyz of the wave's point (16 nbrs x 3)
  __shared__ float sw12[192];
  __shared__ float sb12[64];

  int tid = threadIdx.x;
  int wave = tid >> 5, lane = tid & 31, g = lane >> 4, c = lane & 15;
  int n = blockIdx.x * 8 + wave;

  // Kick off the async weight DMA first, overlap with the scalar prologue.
  stage_async(w3a, sw3a, 8192, tid, 256);
  stage_async(w3b, sw3b, 8192, tid, 256);

  if (tid < 192) sw12[tid] = w12[tid];
  else           sb12[tid - 192] = b12[tid - 192];

  const float* dsrc = dxyz + (size_t)n * 48;
  dz[wave][lane] = dsrc[lane];
  if (lane < 16) dz[wave][32 + lane] = dsrc[32 + lane];

  stage_wait();   // s_wait_asynccnt 0 + barrier

  // p_local[m] = max_k (b1[m] + dxyz_k . w1[:,m]) ; 2 cols per lane
  #pragma unroll
  for (int t = 0; t < 2; ++t){
    int m = 2 * lane + t;
    float wa = w1[m], wb = w1[64 + m], wc = w1[128 + m], bb = b1[m];
    float best = -3.4e38f;
    #pragma unroll
    for (int k = 0; k < 16; ++k){
      float p0 = bb + dz[wave][k*3+0]*wa + dz[wave][k*3+1]*wb + dz[wave][k*3+2]*wc;
      best = fmaxf(best, p0);
    }
    pl[wave][m] = best;
  }
  __syncthreads();

  // Build A operand (pf tile) directly in registers.
  float d0 = dz[wave][c*3+0], d1 = dz[wave][c*3+1], d2 = dz[wave][c*3+2];
  AU A[4];
  #pragma unroll
  for (int kc = 0; kc < 2; ++kc)
    #pragma unroll
    for (int v = 0; v < 8; ++v)
      #pragma unroll
      for (int hf = 0; hf < 2; ++hf){
        int m = kc*32 + ((v>>2)<<4) + (g<<3) + ((v&3)<<1) + hf;  // 0..63
        float val = sb12[m] + d0*sw12[m] + d1*sw12[64+m] + d2*sw12[128+m];
        A[kc].e[2*v + hf] = (_Float16)val;
      }
  #pragma unroll
  for (int kc = 2; kc < 4; ++kc)
    #pragma unroll
    for (int v = 0; v < 8; ++v)
      #pragma unroll
      for (int hf = 0; hf < 2; ++hf){
        int m = (kc-2)*32 + ((v>>2)<<4) + (g<<3) + ((v&3)<<1) + hf;
        A[kc].e[2*v + hf] = (_Float16)pl[wave][m];
      }

  // Prefetch the gathered xp rows while the GEMMs run.
  int kn[8];
  #pragma unroll
  for (int v = 0; v < 8; ++v){
    kn[v] = knn[n * KNN + v + 8*g];
    __builtin_prefetch(&xp[(size_t)kn[v] * DIM + c], 0, 1);
  }

  // GEMM1: h = pf @ w3a + b3a
  v8f acc[8];
  #pragma unroll
  for (int j = 0; j < 8; ++j) acc[j] = splat8(b3a[16*j + c]);
  #pragma unroll
  for (int kc = 0; kc < 4; ++kc)
    gemm8(A[kc].h, sw3a, kc, 0, 8, lane, acc);

  // GELU + stage (C layout -> A layout via LDS)
  #pragma unroll
  for (int j = 0; j < 8; ++j)
    #pragma unroll
    for (int v = 0; v < 8; ++v)
      stg[wave][(v + 8*g)*136 + 16*j + c] = (_Float16)gelu_f(acc[j][v]);
  __syncthreads();

  // GEMM2: pe = h @ w3b + b3b  (preload all A chunks, then pure WMMA stream)
  v16h a2[4];
  #pragma unroll
  for (int kc = 0; kc < 4; ++kc) a2[kc] = loadA_stage(&stg[wave][0], kc, lane);
  v8f acc2[8];
  #pragma unroll
  for (int j = 0; j < 8; ++j) acc2[j] = splat8(b3b[16*j + c]);
  #pragma unroll
  for (int kc = 0; kc < 4; ++kc)
    gemm8(a2[kc], sw3b, kc, 0, 8, lane, acc2);

  // xs + max over the 16 neighbor rows, then BN + residual into feat.
  #pragma unroll
  for (int j = 0; j < 8; ++j){
    int col = 16*j + c;
    float xpc = xp[(size_t)n * DIM + col];
    float mx = -3.4e38f;
    #pragma unroll
    for (int v = 0; v < 8; ++v)
      mx = fmaxf(mx, acc2[j][v] + xp[(size_t)kn[v] * DIM + col] - xpc);
    mx = fmaxf(mx, __shfl_xor(mx, 16, 32));
    if (g == 0)
      feat[(size_t)n * DIM + col] += mx * gamma[col] * BN_INV + beta[col];
  }
}

// ---------------------------------------------------------------------------
// out = BN(feat) @ pp_w   ([N,128]x[128,256])
// ---------------------------------------------------------------------------
__global__ __launch_bounds__(256) void post_kernel(const float* __restrict__ feat,
                                                   const float* __restrict__ pg,
                                                   const float* __restrict__ pb,
                                                   const uint32_t* __restrict__ wp,
                                                   float* __restrict__ out)
{
  __shared__ float sS[128], sT[128];
  if (threadIdx.x < 128){ sS[threadIdx.x] = pg[threadIdx.x] * BN_INV; sT[threadIdx.x] = pb[threadIdx.x]; }
  __syncthreads();
  int wave = threadIdx.x >> 5, lane = threadIdx.x & 31, g = lane >> 4, c = lane & 15;
  int row0 = (blockIdx.x * 8 + wave) * 16;
  const float* rowp = feat + (size_t)(row0 + c) * DIM;
  for (int half = 0; half < 2; ++half){
    v8f acc[8];
    #pragma unroll
    for (int j = 0; j < 8; ++j) acc[j] = splat8(0.f);
    #pragma unroll
    for (int kc = 0; kc < 4; ++kc){
      v16h a = loadA_bn(rowp, sS, sT, kc, g);
      gemm8(a, wp, kc, half*8, 16, lane, acc);
    }
    #pragma unroll
    for (int j = 0; j < 8; ++j)
      #pragma unroll
      for (int v = 0; v < 8; ++v)
        out[(size_t)(row0 + v + 8*g) * 256 + half*128 + 16*j + c] = acc[j][v];
  }
}

// ---------------------------------------------------------------------------
// Host orchestration
// ---------------------------------------------------------------------------
extern "C" void kernel_launch(void* const* d_in, const int* in_sizes, int n_in,
                              void* d_out, int out_size, void* d_ws, size_t ws_size,
                              hipStream_t stream) {
  (void)in_sizes; (void)n_in; (void)out_size; (void)ws_size;
  const float* x       = (const float*)d_in[0];
  const float* xyz     = (const float*)d_in[1];
  const float* g_pos   = (const float*)d_in[2];
  const float* ne_w1   = (const float*)d_in[3];
  const float* ne_g1   = (const float*)d_in[4];
  const float* ne_b1   = (const float*)d_in[5];
  const float* ne_w2   = (const float*)d_in[6];
  const float* ne_g2   = (const float*)d_in[7];
  const float* ne_b2   = (const float*)d_in[8];
  const float* ne_w3   = (const float*)d_in[9];
  const float* nbr_g   = (const float*)d_in[10];
  const float* nbr_b   = (const float*)d_in[11];
  const float* gpe_w   = (const float*)d_in[12];
  const float* bm_w1   = (const float*)d_in[13];
  const float* bm_b1   = (const float*)d_in[14];
  const float* bm_w2   = (const float*)d_in[15];
  const float* bm_g    = (const float*)d_in[16];
  const float* bm_b    = (const float*)d_in[17];
  const float* lfa_proj= (const float*)d_in[18];
  const float* lfa_g   = (const float*)d_in[19];
  const float* lfa_b   = (const float*)d_in[20];
  const float* nca_w1  = (const float*)d_in[21];
  const float* nca_b1  = (const float*)d_in[22];
  const float* nca_w2  = (const float*)d_in[23];
  const float* nca_b2  = (const float*)d_in[24];
  const float* nca_w3a = (const float*)d_in[25];
  const float* nca_b3a = (const float*)d_in[26];
  const float* nca_w3b = (const float*)d_in[27];
  const float* nca_b3b = (const float*)d_in[28];
  const float* m_w1    = (const float*)d_in[29];
  const float* m_b1    = (const float*)d_in[30];
  const float* m_w2    = (const float*)d_in[31];
  const float* m_g     = (const float*)d_in[32];
  const float* m_b     = (const float*)d_in[33];
  const float* pp_g    = (const float*)d_in[34];
  const float* pp_b    = (const float*)d_in[35];
  const float* pp_w    = (const float*)d_in[36];
  const int*   knn     = (const int*)d_in[37];
  float* out = (float*)d_out;

  // Workspace layout
  char* ws = (char*)d_ws;
  float* feat = (float*)(ws + 0);                       // 16 MB
  float* xp   = (float*)(ws + 16777216);                // 16 MB
  float* dxyz = (float*)(ws + 33554432);                // 6.29 MB
  float* w12  = (float*)(ws + 39845888);                // 4*192 f32
  float* b12  = (float*)(ws + 39848960);                // 4*64 f32
  uint32_t* pack = (uint32_t*)(ws + 39849984);          // repacked f16 weights
  uint32_t* p_gpe  = pack;                              // 4096 dw
  uint32_t* p_proj = p_gpe  + 4096;                     // 4 x 8192
  uint32_t* p_w3a  = p_proj + 4*8192;
  uint32_t* p_w3b  = p_w3a  + 4*8192;
  uint32_t* p_bm1  = p_w3b  + 4*8192;                   // 16384
  uint32_t* p_bm2  = p_bm1  + 16384;                    // 16384
  uint32_t* p_m1   = p_bm2  + 16384;                    // 2 x 16384
  uint32_t* p_m2   = p_m1   + 2*16384;
  uint32_t* p_pp   = p_m2   + 2*16384;                  // 16384

  auto rp = [&](const float* W, uint32_t* o, int Kd, int Nd){
    int total = (Kd * Nd) >> 1;
    repack_b16<<<dim3((total + 255)/256), dim3(256), 0, stream>>>(W, o, Kd, Nd);
  };
  rp(gpe_w, p_gpe, 64, 128);
  for (int i = 0; i < 4; ++i){
    rp(lfa_proj + i*16384, p_proj + i*8192, 128, 128);
    rp(nca_w3a  + i*16384, p_w3a  + i*8192, 128, 128);
    rp(nca_w3b  + i*16384, p_w3b  + i*8192, 128, 128);
  }
  rp(bm_w1, p_bm1, 128, 256);
  rp(bm_w2, p_bm2, 256, 128);
  for (int j = 0; j < 2; ++j){
    rp(m_w1 + j*32768, p_m1 + j*16384, 128, 256);
    rp(m_w2 + j*32768, p_m2 + j*16384, 256, 128);
  }
  rp(pp_w, p_pp, 128, 256);
  fuse_w12<<<dim3(1), dim3(256), 0, stream>>>(nca_w1, nca_b1, nca_w2, nca_b2, w12, b12);

  // Stage 1: neighbor embedding
  nbr_kernel<<<dim3(NPTS/256), dim3(256), 0, stream>>>(
      x, xyz, knn, ne_w1, ne_g1, ne_b1, ne_w2, ne_g2, ne_b2, ne_w3, nbr_g, nbr_b, feat, dxyz);

  // Block prologue
  gpe_kernel<<<dim3(NPTS/128), dim3(256), 0, stream>>>(g_pos, p_gpe, feat);
  mlp_kernel<<<dim3(NPTS/128), dim3(256), 0, stream>>>(feat, p_bm1, bm_b1, p_bm2, bm_g, bm_b);

  // 4 LFA layers
  for (int i = 0; i < 4; ++i){
    xp_kernel<<<dim3(NPTS/128), dim3(256), 32768, stream>>>(feat, p_proj + i*8192, xp);
    lfa_kernel<<<dim3(NPTS/8), dim3(256), 65536, stream>>>(
        feat, xp, dxyz, knn,
        nca_w1 + i*192, nca_b1 + i*64, w12 + i*192, b12 + i*64,
        p_w3a + i*8192, nca_b3a + i*128, p_w3b + i*8192, nca_b3b + i*128,
        lfa_g + i*128, lfa_b + i*128);
    if (i & 1){
      int j = i >> 1;
      mlp_kernel<<<dim3(NPTS/128), dim3(256), 0, stream>>>(
          feat, p_m1 + j*16384, m_b1 + j*256, p_m2 + j*16384, m_g + j*128, m_b + j*128);
    }
  }

  // postproj
  post_kernel<<<dim3(NPTS/128), dim3(256), 0, stream>>>(feat, pp_g, pp_b, p_pp, out);
}
